// PhasorBlock_85383949844648
// MI455X (gfx1250) — matrix operand order repaired and checked by reading, not compiled, not run on hardware
//
#include <hip/hip_runtime.h>
#include <math.h>

// ---------------- problem constants ----------------
#define B_   4
#define L_   4096
#define D_   512
#define P_   128
#define V_   8
#define M_   (B_ * L_)        // 16384 tokens
#define NCH  64               // scan chunks
#define CL   64               // chunk length (L_/NCH)
#define PI_F 3.14159265358979323846f

typedef __attribute__((ext_vector_type(16))) _Float16 v16h;
typedef __attribute__((ext_vector_type(8)))  float    v8f;

__device__ __forceinline__ float sigmoidf_(float x) { return 1.f / (1.f + __expf(-x)); }
__device__ __forceinline__ float gelu_(float x)     { return 0.5f * x * (1.f + erff(x * 0.70710678118654752f)); }

enum { EPI_NONE = 0, EPI_SIGMOID_SCALE = 2, EPI_TANH_PI = 3, EPI_ADD_PHI = 4, EPI_GELU = 5, EPI_RESID = 6 };

// =====================================================================
// Weight pre-conversion: fp32 row-major W[Kw, NSRC] -> f16 fragments in
// WMMA B-operand order (cdna5 05_wmma §7.12.2):
//   fragment (ks, ntile): lane l holds col ntile*16+(l&15),
//   K = ks*32 + (l>>4)*8 + i (i<8), ks*32 + 16 + (l>>4)*8 + (i-8) (i>=8).
// Stored at Wf[ ((ks*NT + ntile)*32 + lane)*16 + i ]  (NT = NPAD/16)
// =====================================================================
template<int Kw, int NSRC, int NPAD>
__global__ __launch_bounds__(256) void convw_kernel(const float* __restrict__ W,
                                                    _Float16* __restrict__ Wf)
{
    constexpr int NT = NPAD / 16;
    const int total = (Kw / 32) * NT * 32;
    int g = blockIdx.x * 256 + threadIdx.x;
    if (g >= total) return;
    int lane = g & 31;
    int rest = g >> 5;
    int nt   = rest % NT;
    int ks   = rest / NT;
    int col  = nt * 16 + (lane & 15);
    int kb   = ks * 32 + (lane >> 4) * 8;
    _Float16* dst = Wf + (size_t)g * 16;
#pragma unroll
    for (int i = 0; i < 16; i++)
        dst[i] = (col < NSRC) ? (_Float16)W[(size_t)(kb + i + (i >= 8 ? 8 : 0)) * NSRC + col] : (_Float16)0.f;
}

// =====================================================================
// Activation pre-conversion: fp32 row-major A[M_, Kw] -> f16 fragments in
// WMMA A-operand order: fragment (mt, ks): lane l holds row mt*16+(l&15),
//   K = ks*32 + (l>>4)*8 + {0..7} and + 16 + {0..7}.
// Stored at Af[ ((mt*(Kw/32) + ks)*32 + lane)*16 + i ]
// =====================================================================
template<int Kw>
__global__ __launch_bounds__(256) void conva_kernel(const float* __restrict__ A,
                                                    _Float16* __restrict__ Af)
{
    constexpr int KS = Kw / 32;
    const int total = (M_ / 16) * KS * 32;
    int g = blockIdx.x * 256 + threadIdx.x;
    if (g >= total) return;
    int lane = g & 31;
    int rest = g >> 5;
    int ks   = rest % KS;
    int mt   = rest / KS;
    int row  = mt * 16 + (lane & 15);
    int kb   = ks * 32 + (lane >> 4) * 8;
    const float* src = A + (size_t)row * Kw + kb;
    _Float16* dst = Af + (size_t)g * 16;
#pragma unroll
    for (int i = 0; i < 8; i++) {
        dst[i]     = (_Float16)src[i];
        dst[8 + i] = (_Float16)src[16 + i];
    }
}

// =====================================================================
// WMMA GEMM: C[M_, NSRC] = act( A[M_, K] * W[K, NSRC] + bias )
//  - AF0: A0 is pre-fragmented f16 (zero-cvt path); else fp32 cvt on the fly
//  - A1 (concat tail, K0..K) always fp32 row-major
//  - one wave per 16 x (NTU_*16) strip; A fragment shared by NTU_ tiles
// =====================================================================
template<int K, int K0, int N, int NSRC, int EPI, bool AF0>
__global__ __launch_bounds__(256) void gemm_wmma(
    const float* __restrict__ A0, const _Float16* __restrict__ A0f,
    const float* __restrict__ A1, int sA,
    const _Float16* __restrict__ Wf, const float* __restrict__ bias,
    float* __restrict__ C, int ldc,
    const float* __restrict__ aux, const float* __restrict__ scalep)
{
    constexpr int NT   = N / 16;
    constexpr int NTU_ = (NT >= 8) ? 8 : ((NT >= 4) ? 4 : NT);
    constexpr int NGRP = NT / NTU_;
    constexpr size_t WSTEP = (size_t)NT * 512;      // f16 elems per k-step across all tiles

    const int lane = threadIdx.x & 31;
    const int wave = threadIdx.x >> 5;
    const int tile = blockIdx.x * 8 + wave;
    if (tile >= (M_ / 16) * NGRP) return;           // wave-uniform: EXEC stays all-1 for WMMA
    const int mt = tile / NGRP;
    const int ng = tile % NGRP;
    const int m0 = mt * 16;
    const int nt0 = ng * NTU_;
    const int colC = lane & 15;
    const int rowg = lane >> 4;

    v8f acc[NTU_];
#pragma unroll
    for (int u = 0; u < NTU_; u++) {
        int nc = nt0 * 16 + u * 16 + colC;
        float bv;
        if constexpr (NSRC % 16 != 0) bv = (nc < NSRC) ? bias[nc] : 0.f;
        else                          bv = bias[nc];
#pragma unroll
        for (int r = 0; r < 8; r++) acc[u][r] = bv;
    }

    // A frag: lane holds row lane&15; lanes 0-15 -> K 0..7 & 16..23, lanes 16-31 -> K 8..15 & 24..31
    const int ra  = m0 + (lane & 15);
    const int ka0 = (lane >> 4) * 8;

    const _Float16* Wp = Wf + ((size_t)nt0 * 32 + lane) * 16;

    if constexpr (AF0) {
        const _Float16* Afp = A0f + ((size_t)mt * (K0 / 32) * 32 + lane) * 16;
#pragma unroll 4
        for (int kk = 0; kk < K0; kk += 32) {
            v16h a = *reinterpret_cast<const v16h*>(Afp);
#pragma unroll
            for (int u = 0; u < NTU_; u++) {
                v16h bf = *reinterpret_cast<const v16h*>(Wp + (size_t)u * 512);
                acc[u] = __builtin_amdgcn_wmma_f32_16x16x32_f16(
                    false, a, false, bf, (short)0, acc[u], false, false);
            }
            Afp += 512; Wp += WSTEP;
        }
    } else {
        const float* Ap = A0 + (size_t)ra * sA + ka0;
#pragma unroll 4
        for (int kk = 0; kk < K0; kk += 32) {
            v16h a;
#pragma unroll
            for (int i = 0; i < 8; i++) { a[i] = (_Float16)Ap[i]; a[8 + i] = (_Float16)Ap[16 + i]; }
#pragma unroll
            for (int u = 0; u < NTU_; u++) {
                v16h bf = *reinterpret_cast<const v16h*>(Wp + (size_t)u * 512);
                acc[u] = __builtin_amdgcn_wmma_f32_16x16x32_f16(
                    false, a, false, bf, (short)0, acc[u], false, false);
            }
            Ap += 32; Wp += WSTEP;
        }
    }
    if constexpr (K > K0) {
        const float* Ap = A1 + (size_t)ra * sA + ka0;
#pragma unroll 4
        for (int kk = K0; kk < K; kk += 32) {
            v16h a;
#pragma unroll
            for (int i = 0; i < 8; i++) { a[i] = (_Float16)Ap[i]; a[8 + i] = (_Float16)Ap[16 + i]; }
#pragma unroll
            for (int u = 0; u < NTU_; u++) {
                v16h bf = *reinterpret_cast<const v16h*>(Wp + (size_t)u * 512);
                acc[u] = __builtin_amdgcn_wmma_f32_16x16x32_f16(
                    false, a, false, bf, (short)0, acc[u], false, false);
            }
            Ap += 32; Wp += WSTEP;
        }
    }

    float scale = 1.f;
    if constexpr (EPI == EPI_SIGMOID_SCALE) scale = fabsf(scalep[0]);
#pragma unroll
    for (int u = 0; u < NTU_; u++) {
        int nc = nt0 * 16 + u * 16 + colC;
        if constexpr (NSRC % 16 != 0) { if (nc >= NSRC) continue; }
#pragma unroll
        for (int r = 0; r < 8; r++) {
            int row = m0 + r + rowg * 8;
            float v = acc[u][r];
            if constexpr (EPI == EPI_SIGMOID_SCALE) v = sigmoidf_(v) * scale;
            if constexpr (EPI == EPI_TANH_PI)       v = tanhf(v) * PI_F;
            if constexpr (EPI == EPI_ADD_PHI)       v += aux[(size_t)(row & (L_ - 1)) * D_ + nc];
            if constexpr (EPI == EPI_GELU)          v = gelu_(v);
            if constexpr (EPI == EPI_RESID)         v += aux[(size_t)row * D_ + nc];
            C[(size_t)row * ldc + nc] = v;
        }
    }
}

// =====================================================================
// store_gate: one wave per token, 512-dot with w_g, sigmoid
// =====================================================================
__global__ __launch_bounds__(256) void gate_kernel(const float* __restrict__ x,
                                                   const float* __restrict__ wg,
                                                   const float* __restrict__ bg,
                                                   float* __restrict__ gate)
{
    int wave = threadIdx.x >> 5, lane = threadIdx.x & 31;
    int tok = blockIdx.x * 8 + wave;
    float acc = 0.f;
#pragma unroll
    for (int i = 0; i < D_ / 32; i++) acc += x[(size_t)tok * D_ + lane + 32 * i] * wg[lane + 32 * i];
    for (int off = 16; off; off >>= 1) acc += __shfl_xor(acc, off, 32);
    if (lane == 0) gate[tok] = sigmoidf_(acc + bg[0]);
}

// =====================================================================
// Scan #1 (per-channel): cumsum of mag*v1*cos(phi), *sin(phi), mag, x
// =====================================================================
__global__ __launch_bounds__(512) void scan1_partial(
    const float* __restrict__ v1, const float* __restrict__ mag,
    const float* __restrict__ x,  const float* __restrict__ phases,
    float* __restrict__ pc, float* __restrict__ ps,
    float* __restrict__ pm, float* __restrict__ px)
{
    int b = blockIdx.x / NCH, ch = blockIdx.x % NCH, d = threadIdx.x;
    float sc = 0, ss = 0, sm = 0, sx = 0;
    for (int t = 0; t < CL; t++) {
        int l = ch * CL + t;
        size_t tok = (size_t)b * L_ + l;
        float m = mag[tok * D_ + d], v = v1[tok * D_ + d];
        float ph = phases[(size_t)l * D_ + d];
        float w = m * v;
        sc += w * cosf(ph); ss += w * sinf(ph);
        sm += m;            sx += x[tok * D_ + d];
    }
    size_t o = (size_t)(b * NCH + ch) * D_ + d;
    pc[o] = sc; ps[o] = ss; pm[o] = sm; px[o] = sx;
}

__global__ __launch_bounds__(512) void scan1_prefix(
    float* __restrict__ pc, float* __restrict__ ps,
    float* __restrict__ pm, float* __restrict__ px)
{
    int b = blockIdx.x, d = threadIdx.x;
    float rc = 0, rs = 0, rm = 0, rx = 0;
    for (int ch = 0; ch < NCH; ch++) {
        size_t o = (size_t)(b * NCH + ch) * D_ + d;
        float tc = pc[o]; pc[o] = rc; rc += tc;
        float ts = ps[o]; ps[o] = rs; rs += ts;
        float tm = pm[o]; pm[o] = rm; rm += tm;
        float tx = px[o]; px[o] = rx; rx += tx;
    }
}

__global__ __launch_bounds__(512) void scan1_final(
    const float* __restrict__ v1, const float* __restrict__ mag,
    const float* __restrict__ x,  const float* __restrict__ phases,
    const float* __restrict__ qbuf,
    const float* __restrict__ pc, const float* __restrict__ ps,
    const float* __restrict__ pm, const float* __restrict__ px,
    float* __restrict__ pret, float* __restrict__ cavg)
{
    int b = blockIdx.x / NCH, ch = blockIdx.x % NCH, d = threadIdx.x;
    size_t o = (size_t)(b * NCH + ch) * D_ + d;
    float rc = pc[o], rs = ps[o], rm = pm[o], rx = px[o];
    const float inv_sqrt_d = rsqrtf((float)D_);
    for (int t = 0; t < CL; t++) {
        int l = ch * CL + t;
        size_t tok = (size_t)b * L_ + l;
        float m = mag[tok * D_ + d], v = v1[tok * D_ + d];
        float ph = phases[(size_t)l * D_ + d];
        float w = m * v;
        rc += w * cosf(ph); rs += w * sinf(ph);
        rm += m;            rx += x[tok * D_ + d];
        float sqm = sqrtf(rm + 1e-8f);
        float mc = rc / sqm, msn = rs / sqm;
        float pq = qbuf[tok * D_ + d];
        pret[tok * D_ + d] = (mc * cosf(pq) + msn * sinf(pq)) * inv_sqrt_d;
        cavg[tok * D_ + d] = rx / (float)(l + 1);
    }
}

// =====================================================================
// Scan #2 (KV phasor memory): thread t owns 4 states: p=t>>1, v=(t&1)*4..+3
// =====================================================================
__global__ __launch_bounds__(256) void kv_partial(
    const float* __restrict__ sphase, const float* __restrict__ vals,
    const float* __restrict__ gate,
    float* __restrict__ kpc, float* __restrict__ kps, float* __restrict__ kpg)
{
    __shared__ float lds_sp[P_];
    __shared__ float lds_gv[V_];
    int b = blockIdx.x / NCH, ch = blockIdx.x % NCH, tid = threadIdx.x;
    int p = tid >> 1, vbase = (tid & 1) * 4;
    float sc[4] = {0, 0, 0, 0}, ss[4] = {0, 0, 0, 0}, sg = 0;
    for (int t = 0; t < CL; t++) {
        size_t tok = (size_t)b * L_ + ch * CL + t;
        __syncthreads();
        if (tid < P_) lds_sp[tid] = sphase[tok * P_ + tid];
        if (tid < V_) lds_gv[tid] = vals[tok * V_ + tid] * gate[tok];
        __syncthreads();
        float c = cosf(lds_sp[p]), s = sinf(lds_sp[p]);
#pragma unroll
        for (int j = 0; j < 4; j++) {
            float g = lds_gv[vbase + j];
            sc[j] += c * g; ss[j] += s * g;
        }
        if (tid == 0) sg += gate[tok];
    }
    size_t o = (size_t)(b * NCH + ch) * (P_ * V_) + tid * 4;
#pragma unroll
    for (int j = 0; j < 4; j++) { kpc[o + j] = sc[j]; kps[o + j] = ss[j]; }
    if (tid == 0) kpg[b * NCH + ch] = sg;
}

__global__ __launch_bounds__(256) void kv_prefix(
    float* __restrict__ kpc, float* __restrict__ kps, float* __restrict__ kpg)
{
    int b = blockIdx.x, tid = threadIdx.x;
    float rc[4] = {0, 0, 0, 0}, rs[4] = {0, 0, 0, 0};
    for (int ch = 0; ch < NCH; ch++) {
        size_t o = (size_t)(b * NCH + ch) * (P_ * V_) + tid * 4;
#pragma unroll
        for (int j = 0; j < 4; j++) {
            float tc = kpc[o + j]; kpc[o + j] = rc[j]; rc[j] += tc;
            float ts = kps[o + j]; kps[o + j] = rs[j]; rs[j] += ts;
        }
    }
    if (tid == 0) {
        float rg = 0;
        for (int ch = 0; ch < NCH; ch++) { float tg = kpg[b * NCH + ch]; kpg[b * NCH + ch] = rg; rg += tg; }
    }
}

__global__ __launch_bounds__(256) void kv_final(
    const float* __restrict__ sphase, const float* __restrict__ qpbuf,
    const float* __restrict__ vals,   const float* __restrict__ gate,
    const float* __restrict__ kpc, const float* __restrict__ kps, const float* __restrict__ kpg,
    float* __restrict__ kvret)
{
    __shared__ float lds_sp[P_];
    __shared__ float lds_qp[P_];
    __shared__ float lds_gv[V_];
    __shared__ float lds_red[8 * 8];          // [wave][v]
    int b = blockIdx.x / NCH, ch = blockIdx.x % NCH, tid = threadIdx.x;
    int wave = tid >> 5;
    int p = tid >> 1, vg = tid & 1, vbase = vg * 4;
    size_t o = (size_t)(b * NCH + ch) * (P_ * V_) + tid * 4;
    float rc[4], rs[4];
#pragma unroll
    for (int j = 0; j < 4; j++) { rc[j] = kpc[o + j]; rs[j] = kps[o + j]; }
    float grun = kpg[b * NCH + ch];
    const float inv_sqrt_p = rsqrtf((float)P_);
    for (int t = 0; t < CL; t++) {
        size_t tok = (size_t)b * L_ + ch * CL + t;
        __syncthreads();
        if (tid < P_)      lds_sp[tid]      = sphase[tok * P_ + tid];
        else               lds_qp[tid - P_] = qpbuf[tok * P_ + (tid - P_)];
        if (tid < V_)      lds_gv[tid]      = vals[tok * V_ + tid] * gate[tok];
        float gate_t = gate[tok];
        __syncthreads();
        float c = cosf(lds_sp[p]), s = sinf(lds_sp[p]);
        float cq = cosf(lds_qp[p]), sq = sinf(lds_qp[p]);
        float r[4];
#pragma unroll
        for (int j = 0; j < 4; j++) {
            float g = lds_gv[vbase + j];
            rc[j] += c * g; rs[j] += s * g;
            r[j] = cq * rc[j] + sq * rs[j];
        }
        grun += gate_t;
        float inv = rsqrtf(fmaxf(grun, 1.0f));
        for (int off = 2; off < 32; off <<= 1)
#pragma unroll
            for (int j = 0; j < 4; j++) r[j] += __shfl_xor(r[j], off, 32);
        if ((tid & 31) < 2)
#pragma unroll
            for (int j = 0; j < 4; j++) lds_red[wave * 8 + vg * 4 + j] = r[j];
        __syncthreads();
        if (tid < V_) {
            float acc = 0.f;
#pragma unroll
            for (int w = 0; w < 8; w++) acc += lds_red[w * 8 + tid];
            kvret[tok * V_ + tid] = acc * inv * inv_sqrt_p;
        }
    }
}

// =====================================================================
// kv_out: tiny K=8 GEMM as plain bandwidth kernel -> comb[:, 512:1024]
// =====================================================================
__global__ void kvout_kernel(const float* __restrict__ kvret, const float* __restrict__ wkv,
                             const float* __restrict__ bkv, float* __restrict__ comb)
{
    size_t g = (size_t)blockIdx.x * 256 + threadIdx.x;
    if (g >= (size_t)M_ * D_) return;
    int tok = (int)(g / D_), d = (int)(g % D_);
    float acc = bkv[d];
#pragma unroll
    for (int v = 0; v < V_; v++) acc += kvret[(size_t)tok * V_ + v] * wkv[(size_t)v * D_ + d];
    comb[(size_t)tok * (2 * D_) + D_ + d] = acc;
}

// =====================================================================
// LayerNorm over 1024-wide combined rows, in place (one wave / token)
// =====================================================================
__global__ __launch_bounds__(256) void ln_kernel(float* __restrict__ comb,
                                                 const float* __restrict__ g,
                                                 const float* __restrict__ bta)
{
    int wave = threadIdx.x >> 5, lane = threadIdx.x & 31;
    int tok = blockIdx.x * 8 + wave;
    float vals[32];
    float s = 0, s2 = 0;
#pragma unroll
    for (int i = 0; i < 32; i++) {
        float v = comb[(size_t)tok * (2 * D_) + lane + 32 * i];
        vals[i] = v; s += v; s2 += v * v;
    }
    for (int off = 16; off; off >>= 1) { s += __shfl_xor(s, off, 32); s2 += __shfl_xor(s2, off, 32); }
    float mu = s * (1.f / (2 * D_));
    float var = s2 * (1.f / (2 * D_)) - mu * mu;
    float rstd = rsqrtf(var + 1e-5f);
#pragma unroll
    for (int i = 0; i < 32; i++) {
        int c = lane + 32 * i;
        comb[(size_t)tok * (2 * D_) + c] = (vals[i] - mu) * rstd * g[c] + bta[c];
    }
}

// =====================================================================
// host-side helpers
// =====================================================================
template<int Kw, int NSRC, int NPAD>
static void launch_convw(hipStream_t s, const float* W, _Float16* Wf)
{
    int total = (Kw / 32) * (NPAD / 16) * 32;
    convw_kernel<Kw, NSRC, NPAD><<<(total + 255) / 256, 256, 0, s>>>(W, Wf);
}

template<int K, int K0, int N, int NSRC, int EPI, bool AF0>
static void launch_gemm(hipStream_t s, const float* A0, const _Float16* A0f,
                        const float* A1, int sA,
                        const _Float16* Wf, const float* bias,
                        float* C, int ldc, const float* aux, const float* scalep)
{
    constexpr int NT = N / 16;
    constexpr int NTU_ = (NT >= 8) ? 8 : ((NT >= 4) ? 4 : NT);
    constexpr int NGRP = NT / NTU_;
    int ntiles = (M_ / 16) * NGRP;
    int blocks = (ntiles + 7) / 8;
    gemm_wmma<K, K0, N, NSRC, EPI, AF0><<<blocks, 256, 0, s>>>(A0, A0f, A1, sA, Wf, bias, C, ldc, aux, scalep);
}

extern "C" void kernel_launch(void* const* d_in, const int* in_sizes, int n_in,
                              void* d_out, int out_size, void* d_ws, size_t ws_size,
                              hipStream_t stream)
{
    (void)in_sizes; (void)n_in; (void)out_size; (void)ws_size;
    const float* x      = (const float*)d_in[0];
    const float* phases = (const float*)d_in[1];
    const float* mscale = (const float*)d_in[2];
    const float* w_v  = (const float*)d_in[3];  const float* b_v  = (const float*)d_in[4];
    const float* w_o  = (const float*)d_in[5];  const float* b_o  = (const float*)d_in[6];
    const float* w_m  = (const float*)d_in[7];  const float* b_m  = (const float*)d_in[8];
    const float* w_q  = (const float*)d_in[9];  const float* b_q  = (const float*)d_in[10];
    const float* w_ke = (const float*)d_in[11]; const float* b_ke = (const float*)d_in[12];
    const float* w_ve = (const float*)d_in[13]; const float* b_ve = (const float*)d_in[14];
    const float* w_s1 = (const float*)d_in[15]; const float* b_s1 = (const float*)d_in[16];
    const float* w_s2 = (const float*)d_in[17]; const float* b_s2 = (const float*)d_in[18];
    const float* w_g  = (const float*)d_in[19]; const float* b_g  = (const float*)d_in[20];
    const float* w_kv = (const float*)d_in[21]; const float* b_kv = (const float*)d_in[22];
    const float* ln_g = (const float*)d_in[23]; const float* ln_b = (const float*)d_in[24];
    const float* w_t1 = (const float*)d_in[25]; const float* b_t1 = (const float*)d_in[26];
    const float* w_t2 = (const float*)d_in[27]; const float* b_t2 = (const float*)d_in[28];
    float* out = (float*)d_out;

    // ---- workspace carve-up: f16 region first (32B-aligned base) ----
    _Float16* hw = (_Float16*)d_ws;
    size_t ho = 0;
    _Float16* fw_v  = hw + ho; ho += (size_t)512 * 512;
    _Float16* fw_m  = hw + ho; ho += (size_t)512 * 512;
    _Float16* fw_q  = hw + ho; ho += (size_t)512 * 512;
    _Float16* fw_o  = hw + ho; ho += (size_t)512 * 512;
    _Float16* fw_ke = hw + ho; ho += (size_t)512 * 128;
    _Float16* fw_s2 = hw + ho; ho += (size_t)512 * 128;
    _Float16* fw_ve = hw + ho; ho += (size_t)512 * 16;     // padded 8 -> 16 cols
    _Float16* fw_s1 = hw + ho; ho += (size_t)1024 * 512;
    _Float16* fw_t1 = hw + ho; ho += (size_t)1024 * 1024;
    _Float16* fw_t2 = hw + ho; ho += (size_t)1024 * 512;
    _Float16* fx    = hw + ho; ho += (size_t)M_ * 512;     // x in A-fragment layout

    float* ws = (float*)(hw + ((ho + 15) & ~(size_t)15)); // keep 32B alignment
    size_t off = 0;
    float* v1    = ws + off; off += (size_t)M_ * D_;      // later reused: s1out, then hbuf (with mag)
    float* mag   = ws + off; off += (size_t)M_ * D_;
    float* qbuf  = ws + off; off += (size_t)M_ * D_;      // phi_query
    float* pret  = ws + off; off += (size_t)M_ * D_;      // pos_ret
    float* cavg  = ws + off; off += (size_t)M_ * D_;      // context_avg
    float* comb  = ws + off; off += (size_t)M_ * 2 * D_;  // [pos_out | kv_out], ln in place
    float* qp    = ws + off; off += (size_t)M_ * P_;      // query_phase
    float* sph   = ws + off; off += (size_t)M_ * P_;      // storage_phase
    float* vals  = ws + off; off += (size_t)M_ * V_;
    float* kvr   = ws + off; off += (size_t)M_ * V_;      // kv_retrieved
    float* gate  = ws + off; off += (size_t)M_;
    float* pc    = ws + off; off += (size_t)B_ * NCH * D_;
    float* ps    = ws + off; off += (size_t)B_ * NCH * D_;
    float* pm    = ws + off; off += (size_t)B_ * NCH * D_;
    float* px    = ws + off; off += (size_t)B_ * NCH * D_;
    float* kpc   = ws + off; off += (size_t)B_ * NCH * P_ * V_;
    float* kps   = ws + off; off += (size_t)B_ * NCH * P_ * V_;
    float* kpg   = ws + off; off += (size_t)B_ * NCH;
    float* s1out = v1;
    float* hbuf  = v1;                                    // v1+mag adjacent: M_ x 2D_

    // ---- 0: pre-convert weights (B-frag) and x (A-frag) to f16 ----
    launch_convw< 512, 512,  512>(stream, w_v,  fw_v);
    launch_convw< 512, 512,  512>(stream, w_m,  fw_m);
    launch_convw< 512, 512,  512>(stream, w_q,  fw_q);
    launch_convw< 512, 512,  512>(stream, w_o,  fw_o);
    launch_convw< 512, 128,  128>(stream, w_ke, fw_ke);
    launch_convw< 512, 128,  128>(stream, w_s2, fw_s2);
    launch_convw< 512,   8,   16>(stream, w_ve, fw_ve);
    launch_convw<1024, 512,  512>(stream, w_s1, fw_s1);
    launch_convw<1024, 1024, 1024>(stream, w_t1, fw_t1);
    launch_convw<1024, 512,  512>(stream, w_t2, fw_t2);
    {
        int total = (M_ / 16) * (512 / 32) * 32;
        conva_kernel<512><<<(total + 255) / 256, 256, 0, stream>>>(x, fx);
    }

    // ---- 1..5: token-parallel WMMA projections of x (zero-cvt A path) ----
    launch_gemm<512, 512,  512,  512, EPI_NONE,          true>(stream, x, fx, x, D_, fw_v,  b_v,  v1,   D_, nullptr, nullptr);
    launch_gemm<512, 512,  512,  512, EPI_SIGMOID_SCALE, true>(stream, x, fx, x, D_, fw_m,  b_m,  mag,  D_, nullptr, mscale);
    launch_gemm<512, 512,  512,  512, EPI_ADD_PHI,       true>(stream, x, fx, x, D_, fw_q,  b_q,  qbuf, D_, phases,  nullptr);
    launch_gemm<512, 512,  128,  128, EPI_TANH_PI,       true>(stream, x, fx, x, D_, fw_ke, b_ke, qp,   P_, nullptr, nullptr);
    launch_gemm<512, 512,   16,    8, EPI_NONE,          true>(stream, x, fx, x, D_, fw_ve, b_ve, vals, V_, nullptr, nullptr);
    gate_kernel<<<M_ / 8, 256, 0, stream>>>(x, w_g, b_g, gate);

    // ---- 6..8: chunked scan #1 -> pos_ret, context_avg ----
    scan1_partial<<<B_ * NCH, D_, 0, stream>>>(v1, mag, x, phases, pc, ps, pm, px);
    scan1_prefix <<<B_,       D_, 0, stream>>>(pc, ps, pm, px);
    scan1_final  <<<B_ * NCH, D_, 0, stream>>>(v1, mag, x, phases, qbuf, pc, ps, pm, px, pret, cavg);

    // ---- 9..10: storage-key path (x half pre-fragmented, cavg half raw) ----
    launch_gemm<1024, 512, 512, 512, EPI_GELU,    true >(stream, x, fx, cavg, D_, fw_s1, b_s1, s1out, D_, nullptr, nullptr);
    launch_gemm<512,  512, 128, 128, EPI_TANH_PI, false>(stream, s1out, nullptr, s1out, D_, fw_s2, b_s2, sph, P_, nullptr, nullptr);

    // ---- 11..13: chunked scan #2 -> kv_retrieved ----
    kv_partial<<<B_ * NCH, 256, 0, stream>>>(sph, vals, gate, kpc, kps, kpg);
    kv_prefix <<<B_,       256, 0, stream>>>(kpc, kps, kpg);
    kv_final  <<<B_ * NCH, 256, 0, stream>>>(sph, qp, vals, gate, kpc, kps, kpg, kvr);

    // ---- 14..15: output projections into combined buffer ----
    launch_gemm<512, 512, 512, 512, EPI_NONE, false>(stream, pret, nullptr, pret, D_, fw_o, b_o, comb, 2 * D_, nullptr, nullptr);
    kvout_kernel<<<(int)(((size_t)M_ * D_ + 255) / 256), 256, 0, stream>>>(kvr, w_kv, b_kv, comb);

    // ---- 16: layernorm in place ----
    ln_kernel<<<M_ / 8, 256, 0, stream>>>(comb, ln_g, ln_b);

    // ---- 17..18: output MLP + residual ----
    launch_gemm<1024, 1024, 1024, 1024, EPI_GELU,  false>(stream, comb, nullptr, comb, 2 * D_, fw_t1, b_t1, hbuf, 2 * D_, nullptr, nullptr);
    launch_gemm<1024, 1024,  512,  512, EPI_RESID, false>(stream, hbuf, nullptr, hbuf, 2 * D_, fw_t2, b_t2, out,  D_,     x,       nullptr);
}
